// Loss_5652176961787
// MI455X (gfx1250) — matrix operand order repaired and checked by reading, not compiled, not run on hardware
//
#include <hip/hip_runtime.h>
#include <hip/hip_bf16.h>

#define VOCAB  50257
#define D_EMB  1024
#define N_NODES 4096

typedef __attribute__((ext_vector_type(16))) __bf16 v16bf;
typedef __attribute__((ext_vector_type(8)))  __bf16 v8bf;
typedef __attribute__((ext_vector_type(8)))  float  v8f;

static __device__ __forceinline__ unsigned short f2bf(float f) {
    unsigned u = __float_as_uint(f);
    unsigned r = u + 0x7FFFu + ((u >> 16) & 1u);   // round-to-nearest-even
    return (unsigned short)(r >> 16);
}

#define CMB16(lo, hi) __builtin_shufflevector(lo, hi, \
    0,1,2,3,4,5,6,7,8,9,10,11,12,13,14,15)

// ---------------- Kernel 1: W f32 -> bf16 (workspace) ----------------
__global__ void k_convert_w(const float* __restrict__ W,
                            unsigned short* __restrict__ wbf, int n4) {
    int i = blockIdx.x * blockDim.x + threadIdx.x;
    int stride = gridDim.x * blockDim.x;
    const float4* W4 = (const float4*)W;
    ushort4* O4 = (ushort4*)wbf;
    for (; i < n4; i += stride) {
        float4 v = W4[i];
        ushort4 o;
        o.x = f2bf(v.x); o.y = f2bf(v.y); o.z = f2bf(v.z); o.w = f2bf(v.w);
        O4[i] = o;
    }
}

// ---------------- Kernel 2: target logits t[n] = dot(x[n], W[y[n]]) f32 ----------------
__global__ void k_tgt(const float* __restrict__ x, const float* __restrict__ W,
                      const long long* __restrict__ y, float* __restrict__ t_arr) {
    int n = blockIdx.x * 8 + (threadIdx.x >> 5);
    int lane = threadIdx.x & 31;
    long long tgt = y[n];
    const float4* xr = (const float4*)(x + (size_t)n * D_EMB);
    const float4* wr = (const float4*)(W + (size_t)tgt * D_EMB);
    float acc = 0.f;
    for (int i = lane; i < D_EMB / 4; i += 32) {
        float4 a = xr[i], b = wr[i];
        acc = fmaf(a.x, b.x, acc);
        acc = fmaf(a.y, b.y, acc);
        acc = fmaf(a.z, b.z, acc);
        acc = fmaf(a.w, b.w, acc);
    }
    #pragma unroll
    for (int m = 16; m >= 1; m >>= 1) acc += __shfl_xor(acc, m);
    if (lane == 0) t_arr[n] = acc;
}

// ---------------- Kernel 3: fused GEMM + online logsumexp ----------------
// Block: 256 threads (8 waves), 16 nodes. Each wave owns 4 column tiles per
// 512-col chunk. Explicit 2-stage software pipeline fenced with
// sched_barrier(0): the 8 global + 2 ds loads of stage k+1 are issued and
// kept outstanding while the 4 WMMAs of stage k execute.
#define SX_STRIDE 1032   // 2064 B rows = 129*16 (aligned), dword-stride 516 (conflict-free)

struct Frag {
    v8bf bl[4], bh[4];   // B fragments, 4 column tiles
    v8bf alo, ahi;       // A fragment
};

static __device__ __forceinline__ void load_stage(Frag& f,
        const unsigned short* const wp[4], const unsigned short* ap, int kb) {
    #pragma unroll
    for (int t = 0; t < 4; ++t) {
        f.bl[t] = *(const v8bf*)(wp[t] + kb);
        f.bh[t] = *(const v8bf*)(wp[t] + kb + 16);
    }
    f.alo = *(const v8bf*)(ap + kb);
    f.ahi = *(const v8bf*)(ap + kb + 16);
}

static __device__ __forceinline__ void wmma_stage(v8f c[4], const Frag& f) {
    v16bf A = CMB16(f.alo, f.ahi);
    #pragma unroll
    for (int t = 0; t < 4; ++t) {
        v16bf B = CMB16(f.bl[t], f.bh[t]);
        c[t] = __builtin_amdgcn_wmma_f32_16x16x32_bf16(
            false, A, false, B, (short)0, c[t], false, false);
    }
}

__global__ __launch_bounds__(256)
void k_lse(const float* __restrict__ x, const unsigned short* __restrict__ wbf,
           float* __restrict__ m_arr, float* __restrict__ s_arr) {
    __shared__ unsigned short sx[16 * SX_STRIDE];
    __shared__ float red_m[8][16];
    __shared__ float red_s[8][16];

    const int tid = threadIdx.x;
    const int nb = blockIdx.x * 16;

    // stage x tile to LDS as bf16
    for (int idx = tid; idx < 16 * D_EMB; idx += 256) {
        int r = idx >> 10, k = idx & (D_EMB - 1);
        sx[r * SX_STRIDE + k] = f2bf(x[(size_t)(nb + r) * D_EMB + k]);
    }
    __syncthreads();

    const int wave = tid >> 5;
    const int lane = tid & 31;
    const int r16  = lane & 15;
    const int h    = lane >> 4;

    // per-lane online softmax state: row i (+8 for upper half), lane's column stream
    float m_ln[8], s_ln[8];
    #pragma unroll
    for (int i = 0; i < 8; ++i) { m_ln[i] = -__builtin_inff(); s_ln[i] = 0.f; }

    const unsigned short* ap = sx + r16 * SX_STRIDE + h * 8;

    const int NCH = (VOCAB + 511) / 512;   // 99 chunks of 512 cols
    for (int ch = 0; ch < NCH; ++ch) {
        // scalar chunk guard (wave-uniform by construction)
        int cb = __builtin_amdgcn_readfirstlane(ch * 512 + wave * 64);
        if (cb >= VOCAB) continue;

        const unsigned short* wp[4];
        #pragma unroll
        for (int t = 0; t < 4; ++t) {
            int wr = cb + t * 16 + r16;
            if (wr > VOCAB - 1) wr = VOCAB - 1;
            wp[t] = wbf + (size_t)wr * D_EMB + h * 8;
        }

        // prefetch next chunk's W rows (global_prefetch_b8)
        if (cb + 512 < VOCAB)
            __builtin_prefetch(wbf + (size_t)(cb + 512 + r16) * D_EMB, 0, 1);

        v8f c[4];
        #pragma unroll
        for (int t = 0; t < 4; ++t)
            c[t] = (v8f){0.f, 0.f, 0.f, 0.f, 0.f, 0.f, 0.f, 0.f};

        // ---- 2-stage ping-pong pipeline over K, fenced by sched_barrier ----
        Frag f0, f1;
        load_stage(f0, wp, ap, 0);
        #pragma unroll 1
        for (int kb = 0; kb < D_EMB - 64; kb += 64) {
            load_stage(f1, wp, ap, kb + 32);
            __builtin_amdgcn_sched_barrier(0);
            wmma_stage(c, f0);
            load_stage(f0, wp, ap, kb + 64);
            __builtin_amdgcn_sched_barrier(0);
            wmma_stage(c, f1);
        }
        // kb = D_EMB-64: last two stages
        load_stage(f1, wp, ap, D_EMB - 32);
        __builtin_amdgcn_sched_barrier(0);
        wmma_stage(c, f0);
        __builtin_amdgcn_sched_barrier(0);
        wmma_stage(c, f1);

        // per-lane branchless online update; invalid cols -> -inf -> no-op
        #pragma unroll
        for (int t = 0; t < 4; ++t) {
            bool valid = (cb + t * 16 + r16) < VOCAB;
            #pragma unroll
            for (int i = 0; i < 8; ++i) {
                float v = valid ? c[t][i] : -__builtin_inff();
                float mn = fmaxf(m_ln[i], v);
                s_ln[i] = s_ln[i] * __expf(m_ln[i] - mn) + __expf(v - mn);
                m_ln[i] = mn;
            }
        }
    }

    // one butterfly merge per row across the 16 lanes of each half
    #pragma unroll
    for (int i = 0; i < 8; ++i) {
        float m = m_ln[i], s = s_ln[i];
        #pragma unroll
        for (int mk = 8; mk >= 1; mk >>= 1) {
            float mo = __shfl_xor(m, mk);
            float so = __shfl_xor(s, mk);
            float mn = fmaxf(m, mo);
            s = s * __expf(m - mn) + so * __expf(mo - mn);
            m = mn;
        }
        m_ln[i] = m; s_ln[i] = s;
    }

    // cross-wave merge: lanes 0 and 16 hold rows h*8+i (replicated post-butterfly)
    if (r16 == 0) {
        #pragma unroll
        for (int i = 0; i < 8; ++i) {
            red_m[wave][h * 8 + i] = m_ln[i];
            red_s[wave][h * 8 + i] = s_ln[i];
        }
    }
    __syncthreads();
    if (tid < 16) {
        float m = -__builtin_inff(), s = 0.f;
        #pragma unroll
        for (int w = 0; w < 8; ++w) {
            float mw = red_m[w][tid], sw = red_s[w][tid];
            float mn = fmaxf(m, mw);
            s = s * __expf(m - mn) + sw * __expf(mw - mn);
            m = mn;
        }
        m_arr[nb + tid] = m;
        s_arr[nb + tid] = s;
    }
}

// ---------------- Kernel 4: mean(m + log s - t) ----------------
__global__ void k_final(const float* __restrict__ m_arr, const float* __restrict__ s_arr,
                        const float* __restrict__ t_arr, float* __restrict__ out) {
    __shared__ float red[256];
    float acc = 0.f;
    for (int n = threadIdx.x; n < N_NODES; n += 256)
        acc += m_arr[n] + __logf(s_arr[n]) - t_arr[n];
    red[threadIdx.x] = acc;
    __syncthreads();
    for (int s = 128; s > 0; s >>= 1) {
        if (threadIdx.x < s) red[threadIdx.x] += red[threadIdx.x + s];
        __syncthreads();
    }
    if (threadIdx.x == 0) out[0] = red[0] * (1.0f / N_NODES);
}

extern "C" void kernel_launch(void* const* d_in, const int* in_sizes, int n_in,
                              void* d_out, int out_size, void* d_ws, size_t ws_size,
                              hipStream_t stream) {
    const float* x = (const float*)d_in[0];
    const float* W = (const float*)d_in[1];
    const long long* y = (const long long*)d_in[2];

    unsigned short* wbf = (unsigned short*)d_ws;              // 103 MB bf16 W
    size_t wbf_bytes = (size_t)VOCAB * D_EMB * sizeof(unsigned short);
    float* m_arr = (float*)((char*)d_ws + wbf_bytes);
    float* s_arr = m_arr + N_NODES;
    float* t_arr = s_arr + N_NODES;

    k_convert_w<<<2048, 256, 0, stream>>>(W, wbf, VOCAB * D_EMB / 4);
    k_tgt<<<N_NODES / 8, 256, 0, stream>>>(x, W, y, t_arr);
    k_lse<<<N_NODES / 16, 256, 0, stream>>>(x, wbf, m_arr, s_arr);
    k_final<<<1, 256, 0, stream>>>(m_arr, s_arr, t_arr, (float*)d_out);
}
